// SparseSelfAttention_40398462386785
// MI455X (gfx1250) — matrix operand order repaired and checked by reading, not compiled, not run on hardware
//
#include <hip/hip_runtime.h>

#define BATCH 2
#define SEQ   2048
#define EMB   1024
#define HEADS 16
#define HDIM  64
#define WIN   128
#define TOKENS (BATCH * SEQ)   // 4096
#define KCHUNK 64              // K-slice staged in LDS per pipeline stage

typedef __attribute__((ext_vector_type(16))) __bf16 bf16x16;
typedef __attribute__((ext_vector_type(8)))  float  f32x8;

static __device__ __forceinline__ unsigned short f2bf(float f) {
    unsigned int u = __float_as_uint(f);
    u += 0x7FFFu + ((u >> 16) & 1u);           // round-to-nearest-even
    return (unsigned short)(u >> 16);
}

union FragU { bf16x16 v; uint4 q[2]; };

// A-operand: 16(M) x 32(K) bf16 tile, row-major, ld elements.
static __device__ __forceinline__ bf16x16 load_a_frag(const unsigned short* src, int ld) {
    int lane = threadIdx.x & 31;
    int m = lane & 15, half = lane >> 4;
    const unsigned short* p = src + m * ld + half * 8;
    FragU f;
    f.q[0] = *reinterpret_cast<const uint4*>(p);
    f.q[1] = *reinterpret_cast<const uint4*>(p + 16);
    return f.v;
}

// B-operand from B^T (row-major, rows = N, cols = K): lane holds column N=L%16,
// K = half*8 .. half*8+7 and 16+half*8 .. 16+half*8+7.
static __device__ __forceinline__ bf16x16 load_b_frag(const unsigned short* srcT, int ld) {
    int lane = threadIdx.x & 31;
    int m = lane & 15, half = lane >> 4;
    const unsigned short* p = srcT + m * ld + half * 8;
    FragU f;
    f.q[0] = *reinterpret_cast<const uint4*>(p);
    f.q[1] = *reinterpret_cast<const uint4*>(p + 16);
    return f.v;
}

static __device__ __forceinline__ f32x8 wmma_bf16(bf16x16 a, bf16x16 b, f32x8 c) {
    return __builtin_amdgcn_wmma_f32_16x16x32_bf16(false, a, false, b, (short)0, c, false, false);
}

// CDNA5 async global->LDS copy (ASYNCcnt-tracked, no VGPR round trip).
static __device__ __forceinline__ void async_b128(unsigned lds_off, const void* gaddr) {
    unsigned long long a = (unsigned long long)(uintptr_t)gaddr;
    asm volatile("global_load_async_to_lds_b128 %0, %1, off"
                 :: "v"(lds_off), "v"(a) : "memory");
}
static __device__ __forceinline__ void wait_async0() {
    asm volatile("s_wait_asynccnt 0x0" ::: "memory");
}

// ---------------- conversion kernels ----------------

__global__ void cvt_f32_bf16(const float* __restrict__ in, unsigned short* __restrict__ out, int n) {
    int i = blockIdx.x * blockDim.x + threadIdx.x;
    if (i < n) out[i] = f2bf(in[i]);
}

__global__ void cvt_transpose(const float* __restrict__ in, unsigned short* __restrict__ out) {
    int i = blockIdx.x * blockDim.x + threadIdx.x;   // over EMB*EMB
    if (i >= EMB * EMB) return;
    int nr = i >> 10;
    int kc = i & (EMB - 1);
    out[i] = f2bf(in[kc * EMB + nr]);
}

// ------------- GEMM: block = 256(M) x 128(N), 8 waves x (32M x 128N), B^T in LDS -------------
// Each B-fragment fetched from LDS feeds TWO WMMAs (rows lo/hi) -> halves LDS bytes per FLOP.

static __device__ __forceinline__ void stage_async(unsigned lds_base_bytes,
                                                   const unsigned short* __restrict__ Bt,
                                                   int n0, int kc) {
    // 128 rows x 64 cols bf16 = 16 KB = 1024 chunks of 16 B; 256 threads x 4 chunks.
    int tid = threadIdx.x;
#pragma unroll
    for (int j = 0; j < 4; ++j) {
        int idx = tid + j * 256;          // 0..1023
        int n   = idx >> 3;               // 0..127
        int kq  = idx & 7;                // 8-element chunk within row
        unsigned lds_off = lds_base_bytes + (unsigned)(n * KCHUNK + kq * 8) * 2u;
        const unsigned short* g = Bt + (size_t)(n0 + n) * EMB + kc + kq * 8;
        async_b128(lds_off, g);
    }
}

// acc[0][nt] = rows m0..m0+15, acc[1][nt] = rows m0+16..m0+31
static __device__ __forceinline__ void gemm_core(const unsigned short* __restrict__ Arow,
                                                 const unsigned short* __restrict__ Bt,
                                                 int n0, unsigned short* bstage,
                                                 f32x8 acc[2][8]) {
    const int K = EMB;
    unsigned lds_base = (unsigned)(uintptr_t)bstage;
    const unsigned bufBytes = 128 * KCHUNK * 2;
    const unsigned short* ArowHi = Arow + (size_t)16 * K;

    // prologue: stage chunk 0, prefetch 4 A frags (2 M-halves x 2 K-halves)
    stage_async(lds_base, Bt, n0, 0);
    bf16x16 aL0 = load_a_frag(Arow, K);
    bf16x16 aL1 = load_a_frag(Arow + 32, K);
    bf16x16 aH0 = load_a_frag(ArowHi, K);
    bf16x16 aH1 = load_a_frag(ArowHi + 32, K);
    wait_async0();
    __syncthreads();

    for (int kc = 0; kc < K; kc += KCHUNK) {
        int curb = (kc / KCHUNK) & 1;
        const unsigned short* cur = bstage + (size_t)curb * (128 * KCHUNK);

        bf16x16 aL0n, aL1n, aH0n, aH1n;
        if (kc + KCHUNK < K) {
            stage_async(lds_base + (curb ^ 1) * bufBytes, Bt, n0, kc + KCHUNK);
            aL0n = load_a_frag(Arow + kc + KCHUNK, K);
            aL1n = load_a_frag(Arow + kc + KCHUNK + 32, K);
            aH0n = load_a_frag(ArowHi + kc + KCHUNK, K);
            aH1n = load_a_frag(ArowHi + kc + KCHUNK + 32, K);
        }

#pragma unroll
        for (int kh = 0; kh < 2; ++kh) {
            bf16x16 aL = kh ? aL1 : aL0;
            bf16x16 aH = kh ? aH1 : aH0;
#pragma unroll
            for (int nt = 0; nt < 8; ++nt) {
                bf16x16 b = load_b_frag(cur + (size_t)nt * 16 * KCHUNK + kh * 32, KCHUNK);
                acc[0][nt] = wmma_bf16(aL, b, acc[0][nt]);   // B reused twice per fetch
                acc[1][nt] = wmma_bf16(aH, b, acc[1][nt]);
            }
        }

        aL0 = aL0n; aL1 = aL1n; aH0 = aH0n; aH1 = aH1n;
        wait_async0();       // own async stores into next buffer done
        __syncthreads();     // everyone's stores done before next reads
    }
}

// QKV projection. vmode==0: store [B,H,S,D]; vmode==1: store transposed [B,H,D,S].
__global__ __launch_bounds__(256) void proj_kernel(const unsigned short* __restrict__ xbf,
                                                   const unsigned short* __restrict__ WT,
                                                   unsigned short* __restrict__ dst, int vmode) {
    __shared__ __align__(16) unsigned short bstage[2 * 128 * KCHUNK];
    int bm = blockIdx.x >> 3;           // 16 M-blocks of 256 rows
    int n0 = (blockIdx.x & 7) * 128;    // 8 N-blocks
    int wave = threadIdx.x >> 5;
    int m0 = bm * 256 + wave * 32;

    const f32x8 z = {0, 0, 0, 0, 0, 0, 0, 0};
    f32x8 acc[2][8];
#pragma unroll
    for (int mh = 0; mh < 2; ++mh)
#pragma unroll
        for (int i = 0; i < 8; ++i) acc[mh][i] = z;

    gemm_core(xbf + (size_t)m0 * EMB, WT, n0, bstage, acc);

    int lane = threadIdx.x & 31;
    int mloc = lane & 15, half = lane >> 4;
#pragma unroll
    for (int mh = 0; mh < 2; ++mh) {
#pragma unroll
        for (int nt = 0; nt < 8; ++nt) {
#pragma unroll
            for (int r = 0; r < 8; ++r) {
                int tok = m0 + mh * 16 + r + 8 * half;
                int n = n0 + nt * 16 + mloc;
                unsigned short bv = f2bf(acc[mh][nt][r]);
                int b = tok >> 11, s = tok & (SEQ - 1);
                int h = n >> 6, d = n & (HDIM - 1);
                if (vmode == 0)
                    dst[(((size_t)(b * HEADS + h) * SEQ) + s) * HDIM + d] = bv;
                else
                    dst[(((size_t)(b * HEADS + h) * HDIM) + d) * SEQ + s] = bv;
            }
        }
    }
}

// Output projection with residual: out = attn @ Wo + x (fp32)
__global__ __launch_bounds__(256) void outproj_kernel(const unsigned short* __restrict__ attnbf,
                                                      const unsigned short* __restrict__ WoT,
                                                      const float* __restrict__ x,
                                                      float* __restrict__ out) {
    __shared__ __align__(16) unsigned short bstage[2 * 128 * KCHUNK];
    int bm = blockIdx.x >> 3;
    int n0 = (blockIdx.x & 7) * 128;
    int wave = threadIdx.x >> 5;
    int m0 = bm * 256 + wave * 32;

    const f32x8 z = {0, 0, 0, 0, 0, 0, 0, 0};
    f32x8 acc[2][8];
#pragma unroll
    for (int mh = 0; mh < 2; ++mh)
#pragma unroll
        for (int i = 0; i < 8; ++i) acc[mh][i] = z;

    gemm_core(attnbf + (size_t)m0 * EMB, WoT, n0, bstage, acc);

    int lane = threadIdx.x & 31;
    int mloc = lane & 15, half = lane >> 4;
#pragma unroll
    for (int mh = 0; mh < 2; ++mh) {
#pragma unroll
        for (int nt = 0; nt < 8; ++nt) {
#pragma unroll
            for (int r = 0; r < 8; ++r) {
                int tok = m0 + mh * 16 + r + 8 * half;
                int n = n0 + nt * 16 + mloc;
                size_t idx = (size_t)tok * EMB + n;
                out[idx] = acc[mh][nt][r] + x[idx];
            }
        }
    }
}

// ---------------- sliding-window attention (flash-style) ----------------
__global__ __launch_bounds__(256) void attn_kernel(const unsigned short* __restrict__ qbf,
                                                   const unsigned short* __restrict__ kbf,
                                                   const unsigned short* __restrict__ vT,
                                                   unsigned short* __restrict__ attnbf) {
    __shared__ __align__(16) unsigned short plds[8][16 * 32];

    int wslot = threadIdx.x >> 5;
    int wave = blockIdx.x * 8 + wslot;
    int st = wave & (SEQ / 16 - 1);
    int bh = wave >> 7;
    int s0 = st * 16;

    const unsigned short* qb = qbf + (size_t)bh * SEQ * HDIM;
    const unsigned short* kb = kbf + (size_t)bh * SEQ * HDIM;
    const unsigned short* vb = vT + (size_t)bh * HDIM * SEQ;

    int lane = threadIdx.x & 31;
    int mloc = lane & 15, half = lane >> 4;

    bf16x16 qA0 = load_a_frag(qb + (size_t)s0 * HDIM, HDIM);
    bf16x16 qA1 = load_a_frag(qb + (size_t)s0 * HDIM + 32, HDIM);

    const f32x8 z = {0, 0, 0, 0, 0, 0, 0, 0};
    float mrow[8], lrow[8];
    f32x8 o[4];
#pragma unroll
    for (int r = 0; r < 8; ++r) { mrow[r] = -1e30f; lrow[r] = 0.0f; }
#pragma unroll
    for (int i = 0; i < 4; ++i) o[i] = z;

    int lo = s0 - (WIN - 1);
    int t0 = (lo <= 0) ? 0 : (lo & ~31);
    for (; t0 <= s0 + 15; t0 += 32) {
        // batch ALL loads for this key tile up front: k frags AND v frags.
        bf16x16 kf[4];
        kf[0] = load_b_frag(kb + (size_t)t0 * HDIM, HDIM);
        kf[1] = load_b_frag(kb + (size_t)t0 * HDIM + 32, HDIM);
        kf[2] = load_b_frag(kb + (size_t)(t0 + 16) * HDIM, HDIM);
        kf[3] = load_b_frag(kb + (size_t)(t0 + 16) * HDIM + 32, HDIM);
        bf16x16 vf[4];
#pragma unroll
        for (int i = 0; i < 4; ++i)
            vf[i] = load_b_frag(vb + (size_t)(i * 16) * SEQ + t0, SEQ);

        f32x8 sc0 = z, sc1 = z;
        sc0 = wmma_bf16(qA0, kf[0], sc0);
        sc0 = wmma_bf16(qA1, kf[1], sc0);
        sc1 = wmma_bf16(qA0, kf[2], sc1);
        sc1 = wmma_bf16(qA1, kf[3], sc1);

        // streaming softmax (row stats via 16-lane shfl reductions)
#pragma unroll
        for (int r = 0; r < 8; ++r) {
            int srow = s0 + r + 8 * half;
            int ta = t0 + mloc;
            int tb = t0 + 16 + mloc;
            float x0 = sc0[r] * 0.125f;   // 1/sqrt(64)
            float x1 = sc1[r] * 0.125f;
            if (!(ta <= srow && ta >= srow - (WIN - 1))) x0 = -1e9f;
            if (!(tb <= srow && tb >= srow - (WIN - 1))) x1 = -1e9f;
            float rmax = fmaxf(x0, x1);
#pragma unroll
            for (int off = 1; off < 16; off <<= 1)
                rmax = fmaxf(rmax, __shfl_xor(rmax, off, 32));
            float mnew = fmaxf(mrow[r], rmax);
            float alpha = __expf(mrow[r] - mnew);
            float p0 = __expf(x0 - mnew);
            float p1 = __expf(x1 - mnew);
            float rsum = p0 + p1;
#pragma unroll
            for (int off = 1; off < 16; off <<= 1)
                rsum += __shfl_xor(rsum, off, 32);
            lrow[r] = lrow[r] * alpha + rsum;
            mrow[r] = mnew;
#pragma unroll
            for (int i = 0; i < 4; ++i) o[i][r] *= alpha;
            int lr = r + 8 * half;
            plds[wslot][lr * 32 + mloc] = f2bf(p0);
            plds[wslot][lr * 32 + 16 + mloc] = f2bf(p1);
        }
        // C-layout -> A-layout via LDS (same-wave DS ops are in-order)
        bf16x16 pA = load_a_frag(&plds[wslot][0], 32);
#pragma unroll
        for (int i = 0; i < 4; ++i)
            o[i] = wmma_bf16(pA, vf[i], o[i]);
    }

    int b = bh >> 4, h = bh & 15;
#pragma unroll
    for (int i = 0; i < 4; ++i) {
#pragma unroll
        for (int r = 0; r < 8; ++r) {
            int srow = s0 + r + 8 * half;
            int col = h * HDIM + i * 16 + mloc;
            attnbf[((size_t)(b * SEQ + srow)) * EMB + col] = f2bf(o[i][r] / lrow[r]);
        }
    }
}

// ---------------- launch ----------------

extern "C" void kernel_launch(void* const* d_in, const int* in_sizes, int n_in,
                              void* d_out, int out_size, void* d_ws, size_t ws_size,
                              hipStream_t stream) {
    (void)in_sizes; (void)n_in; (void)out_size; (void)ws_size;
    const float* x  = (const float*)d_in[0];
    const float* Wq = (const float*)d_in[1];
    const float* Wk = (const float*)d_in[2];
    const float* Wv = (const float*)d_in[3];
    const float* Wo = (const float*)d_in[4];

    char* w = (char*)d_ws;
    unsigned short* xbf    = (unsigned short*)w; w += (size_t)TOKENS * EMB * 2;
    unsigned short* WqT    = (unsigned short*)w; w += (size_t)EMB * EMB * 2;
    unsigned short* WkT    = (unsigned short*)w; w += (size_t)EMB * EMB * 2;
    unsigned short* WvT    = (unsigned short*)w; w += (size_t)EMB * EMB * 2;
    unsigned short* WoT    = (unsigned short*)w; w += (size_t)EMB * EMB * 2;
    unsigned short* qbf    = (unsigned short*)w; w += (size_t)TOKENS * EMB * 2;
    unsigned short* kbf    = (unsigned short*)w; w += (size_t)TOKENS * EMB * 2;
    unsigned short* vTbf   = (unsigned short*)w; w += (size_t)TOKENS * EMB * 2;
    unsigned short* attnbf = (unsigned short*)w; w += (size_t)TOKENS * EMB * 2;

    int n = TOKENS * EMB;
    cvt_f32_bf16<<<(n + 255) / 256, 256, 0, stream>>>(x, xbf, n);
    int nw = EMB * EMB;
    cvt_transpose<<<(nw + 255) / 256, 256, 0, stream>>>(Wq, WqT);
    cvt_transpose<<<(nw + 255) / 256, 256, 0, stream>>>(Wk, WkT);
    cvt_transpose<<<(nw + 255) / 256, 256, 0, stream>>>(Wv, WvT);
    cvt_transpose<<<(nw + 255) / 256, 256, 0, stream>>>(Wo, WoT);

    // block = 256x128 output tile; grid = 16 * 8 = 128 blocks (1024 GEMM waves)
    int gemm_blocks = (TOKENS / 256) * (EMB / 128);
    proj_kernel<<<gemm_blocks, 256, 0, stream>>>(xbf, WqT, qbf, 0);
    proj_kernel<<<gemm_blocks, 256, 0, stream>>>(xbf, WkT, kbf, 0);
    proj_kernel<<<gemm_blocks, 256, 0, stream>>>(xbf, WvT, vTbf, 1);

    int attn_blocks = (BATCH * HEADS * (SEQ / 16)) / 8;   // 512
    attn_kernel<<<attn_blocks, 256, 0, stream>>>(qbf, kbf, vTbf, attnbf);

    outproj_kernel<<<gemm_blocks, 256, 0, stream>>>(attnbf, WoT, x, (float*)d_out);
}
